// HardNegativeContrastiveLoss_79250736545851
// MI455X (gfx1250) — matrix operand order repaired and checked by reading, not compiled, not run on hardware
//
#include <hip/hip_runtime.h>
#include <hip/hip_bf16.h>
#include <math.h>

#define INV_TEMP (1.0f / 0.07f)

typedef __attribute__((ext_vector_type(16))) __bf16 v16bf_t;
typedef __attribute__((ext_vector_type(8)))  __bf16 v8bf_t;
typedef __attribute__((ext_vector_type(8)))  float  v8f_t;

union V16U { v16bf_t v; v8bf_t h[2]; };

// ---------------------------------------------------------------------------
// Split fp32 -> bf16 hi/lo (Ootomo-style). Vision side pre-scaled by 1/TEMP.
// ---------------------------------------------------------------------------
__global__ void hnc_convert(const float* __restrict__ v, const float* __restrict__ t,
                            __bf16* __restrict__ vhi, __bf16* __restrict__ vlo,
                            __bf16* __restrict__ thi, __bf16* __restrict__ tlo, int n) {
    int i = blockIdx.x * blockDim.x + threadIdx.x;
    if (i >= n) return;
    float x = v[i] * INV_TEMP;
    __bf16 xh = (__bf16)x;
    vhi[i] = xh;
    vlo[i] = (__bf16)(x - (float)xh);
    float y = t[i];
    __bf16 yh = (__bf16)y;
    thi[i] = yh;
    tlo[i] = (__bf16)(y - (float)yh);
}

// ---------------------------------------------------------------------------
// Exact fp32 diagonal: s_ii = dot(v_i, t_i)/TEMP. One wave32 per row.
// ---------------------------------------------------------------------------
__global__ void hnc_diag(const float* __restrict__ v, const float* __restrict__ t,
                         float* __restrict__ diag, int D) {
    int row  = blockIdx.x * (blockDim.x >> 5) + (threadIdx.x >> 5);
    int lane = threadIdx.x & 31;
    const float* vp = v + (size_t)row * D;
    const float* tp = t + (size_t)row * D;
    float s = 0.f;
    for (int j = lane; j < D; j += 32) s += vp[j] * tp[j];
    for (int off = 16; off > 0; off >>= 1) s += __shfl_down(s, off, 32);
    if (lane == 0) diag[row] = s * INV_TEMP;
}

__global__ void hnc_zero(float* __restrict__ p, int n) {
    int i = blockIdx.x * blockDim.x + threadIdx.x;
    if (i < n) p[i] = 0.f;
}

// ---------------------------------------------------------------------------
// Single fused pass. Block = 128 threads (4 waves), owns 16 rows; each wave
// walks column tiles stride-4. Per 16x16x256 tile: 8 K-steps x 3 bf16 WMMAs
// (hi*hi + hi*lo + lo*hi ~ fp32 accuracy). From each C-fragment we derive:
//   - column exp-sums directly from the column-major C layout (-> LDS
//     colpart[], flushed with global atomics at block end),
//   - row exp-sums + per-row top-4 hard negatives via an LDS-staged
//     row-major re-read of E = exp(sim). top-k on E == top-k on sim, and the
//     alpha=2 correction is E^2 - E, so raw logits are never needed again.
// ---------------------------------------------------------------------------
__global__ __launch_bounds__(128)
void hnc_pass(const __bf16* __restrict__ Ahi, const __bf16* __restrict__ Alo,
              const __bf16* __restrict__ Bhi, const __bf16* __restrict__ Blo,
              float* __restrict__ rowsum, float* __restrict__ colsum,
              float* __restrict__ hcorr, int* __restrict__ hidx, int Bn) {
    __shared__ __align__(32) __bf16 lds_ahi[16][256];
    __shared__ __align__(32) __bf16 lds_alo[16][256];
    __shared__ float stage[4][16][17];   // per-wave E-tile staging (padded)
    __shared__ float colpart[8192];      // block-local column partial sums
    __shared__ float lsum[128];
    __shared__ float ltv[128][4];
    __shared__ int   lti[128][4];

    const int tid     = threadIdx.x;
    const int lane    = tid & 31;
    const int wave    = tid >> 5;
    const int rowbase = blockIdx.x * 16;

    // Cache this block's 16 A-rows (hi+lo) in LDS; zero column partials.
    for (int i = tid; i < 16 * 256; i += 128) {
        int r = i >> 8, k = i & 255;
        lds_ahi[r][k] = Ahi[(size_t)(rowbase + r) * 256 + k];
        lds_alo[r][k] = Alo[(size_t)(rowbase + r) * 256 + k];
    }
    for (int i = tid; i < Bn; i += 128) colpart[i] = 0.f;
    __syncthreads();

    // WMMA operand addressing (contiguous k = 16h + s, 16-bit A 16x32 / B 32x16, wave32)
    const int m      = lane & 15;              // A row within tile
    const int a_koff = (lane < 16) ? 0 : 16;   // A K sub-offset per lane half
    const int b_koff = (lane < 16) ? 0 : 16;   // B K sub-offset per lane half
    const int nloc   = lane & 15;              // B/C column within tile

    float sum = 0.f;                            // row partial (this lane)
    float t0 = 0.f, t1 = 0.f, t2 = 0.f, t3 = 0.f;  // top-4 E values (E > 0)
    int   i0 = 0, i1 = 0, i2 = 0, i3 = 0;

    const int ntiles = Bn >> 4;
    for (int ct = wave; ct < ntiles; ct += 4) {
        const int colbase = ct << 4;
        const __bf16* bh_row = Bhi + (size_t)(colbase + nloc) * 256;
        const __bf16* bl_row = Blo + (size_t)(colbase + nloc) * 256;

        v8f_t c = {0.f, 0.f, 0.f, 0.f, 0.f, 0.f, 0.f, 0.f};
#pragma unroll
        for (int ks = 0; ks < 8; ++ks) {
            const int k0 = ks * 32;
            V16U ah, al;
            ah.h[0] = *(const v8bf_t*)&lds_ahi[m][k0 + a_koff];
            ah.h[1] = *(const v8bf_t*)&lds_ahi[m][k0 + a_koff + 8];
            al.h[0] = *(const v8bf_t*)&lds_alo[m][k0 + a_koff];
            al.h[1] = *(const v8bf_t*)&lds_alo[m][k0 + a_koff + 8];
            v16bf_t bh = *(const v16bf_t*)(bh_row + k0 + b_koff);
            v16bf_t bl = *(const v16bf_t*)(bl_row + k0 + b_koff);
            c = __builtin_amdgcn_wmma_f32_16x16x32_bf16(false, ah.v, false, bh, (short)0, c, false, false);
            c = __builtin_amdgcn_wmma_f32_16x16x32_bf16(false, ah.v, false, bl, (short)0, c, false, false);
            c = __builtin_amdgcn_wmma_f32_16x16x32_bf16(false, al.v, false, bh, (short)0, c, false, false);
        }

        // Exponentiate in C layout: lane holds 8 rows of column colbase+nloc.
        // Stage E to LDS for the row-major pass, accumulate column partial.
        const int mbase = (lane < 16) ? 0 : 8;
        float colacc = 0.f;
#pragma unroll
        for (int e = 0; e < 8; ++e) {
            const float E = __expf(c[e]);
            stage[wave][mbase + e][nloc] = E;
            colacc += E;
        }
        colacc += __shfl_xor(colacc, 16, 32);     // join lanes N and N+16
        if (lane < 16) atomicAdd(&colpart[colbase + lane], colacc);
        asm volatile("" ::: "memory");  // DS ops from one wave are in-order

        // Re-read organized by row: 2 lanes per row, 8 columns each.
        const int r    = lane >> 1;
        const int h    = lane & 1;
        const int rowg = rowbase + r;
#pragma unroll
        for (int j = 0; j < 8; ++j) {
            const int   cg = colbase + h * 8 + j;
            const float E  = stage[wave][r][h * 8 + j];
            sum += E;
            if (cg != rowg && E > t3) {  // diag excluded from top-k only
                if (E > t0)      { t3=t2;i3=i2; t2=t1;i2=i1; t1=t0;i1=i0; t0=E;i0=cg; }
                else if (E > t1) { t3=t2;i3=i2; t2=t1;i2=i1; t1=E;i1=cg; }
                else if (E > t2) { t3=t2;i3=i2; t2=E;i2=cg; }
                else             { t3=E;i3=cg; }
            }
        }
        asm volatile("" ::: "memory");
    }

    lsum[tid] = sum;
    ltv[tid][0]=t0; ltv[tid][1]=t1; ltv[tid][2]=t2; ltv[tid][3]=t3;
    lti[tid][0]=i0; lti[tid][1]=i1; lti[tid][2]=i2; lti[tid][3]=i3;
    __syncthreads();

    // Flush block-local column partials to the global column sums.
    for (int i = tid; i < Bn; i += 128) atomicAdd(&colsum[i], colpart[i]);

    // Final per-row merge: 8 partials per row (4 waves x 2 lanes).
    if (tid < 16) {
        const int r = tid;
        float plain = 0.f;
        for (int w = 0; w < 4; ++w)
            for (int h = 0; h < 2; ++h)
                plain += lsum[w * 32 + r * 2 + h];
        float cv[32]; int ci[32]; int n = 0;
        for (int w = 0; w < 4; ++w)
            for (int h = 0; h < 2; ++h) {
                const int src = w * 32 + r * 2 + h;
                for (int k = 0; k < 4; ++k) { cv[n] = ltv[src][k]; ci[n] = lti[src][k]; ++n; }
            }
        float rs = plain;
        for (int k = 0; k < 4; ++k) {           // select global top-4 of 32
            int best = 0;
            for (int q = 1; q < 32; ++q) if (cv[q] > cv[best]) best = q;
            const float E    = cv[best];
            const float corr = E * E - E;        // e^{2a} - e^{a}  (alpha=2)
            rs += corr;
            hcorr[(size_t)(rowbase + r) * 4 + k] = corr;
            hidx [(size_t)(rowbase + r) * 4 + k] = ci[best];
            cv[best] = -1.f;
        }
        rowsum[rowbase + r] = rs;
    }
}

// ---------------------------------------------------------------------------
// Apply hard-negative corrections to the column exp-sums.
// ---------------------------------------------------------------------------
__global__ void hnc_scatter(const int* __restrict__ hidx, const float* __restrict__ hcorr,
                            float* __restrict__ colsum, int n) {
    int i = blockIdx.x * blockDim.x + threadIdx.x;
    if (i < n) atomicAdd(&colsum[hidx[i]], hcorr[i]);
}

// ---------------------------------------------------------------------------
// loss = (1/2B) * sum_i [ log rowsum_i + log colsum_i - 2*diag_i ]
// ---------------------------------------------------------------------------
__global__ void hnc_reduce(const float* __restrict__ rowsum, const float* __restrict__ colsum,
                           const float* __restrict__ diag, float* __restrict__ out, int Bn) {
    __shared__ float partial[256];
    const int tid = threadIdx.x;
    float acc = 0.f;
    for (int i = tid; i < Bn; i += 256)
        acc += logf(rowsum[i]) + logf(colsum[i]) - 2.f * diag[i];
    partial[tid] = acc;
    __syncthreads();
    for (int s = 128; s > 0; s >>= 1) {
        if (tid < s) partial[tid] += partial[tid + s];
        __syncthreads();
    }
    if (tid == 0) out[0] = partial[0] / (2.f * (float)Bn);
}

extern "C" void kernel_launch(void* const* d_in, const int* in_sizes, int n_in,
                              void* d_out, int out_size, void* d_ws, size_t ws_size,
                              hipStream_t stream) {
    const float* v = (const float*)d_in[0];
    const float* t = (const float*)d_in[1];
    const int D  = 256;
    const int n  = in_sizes[0];       // B*D
    const int Bn = n / D;             // 8192

    char* p = (char*)d_ws;
    auto carve = [&](size_t bytes) -> char* {
        char* r = p;
        p += (bytes + 255) & ~(size_t)255;
        return r;
    };
    __bf16* vhi = (__bf16*)carve((size_t)n * 2);
    __bf16* vlo = (__bf16*)carve((size_t)n * 2);
    __bf16* thi = (__bf16*)carve((size_t)n * 2);
    __bf16* tlo = (__bf16*)carve((size_t)n * 2);
    float* rowsum = (float*)carve((size_t)Bn * 4);
    float* colsum = (float*)carve((size_t)Bn * 4);
    float* diag   = (float*)carve((size_t)Bn * 4);
    float* hcorr  = (float*)carve((size_t)Bn * 4 * 4);
    int*   hidx   = (int*)  carve((size_t)Bn * 4 * 4);

    hnc_convert<<<(n + 255) / 256, 256, 0, stream>>>(v, t, vhi, vlo, thi, tlo, n);
    hnc_diag<<<Bn / 8, 256, 0, stream>>>(v, t, diag, D);
    hnc_zero<<<(Bn + 255) / 256, 256, 0, stream>>>(colsum, Bn);
    // Single fused pass: row sums + top-4 + column sums in one GEMM.
    hnc_pass<<<Bn / 16, 128, 0, stream>>>(vhi, vlo, thi, tlo,
                                          rowsum, colsum, hcorr, hidx, Bn);
    hnc_scatter<<<(Bn * 4 + 255) / 256, 256, 0, stream>>>(hidx, hcorr, colsum, Bn * 4);
    hnc_reduce<<<1, 256, 0, stream>>>(rowsum, colsum, diag, (float*)d_out, Bn);
}
